// PerformerAttention_63402307224070
// MI455X (gfx1250) — matrix operand order repaired and checked by reading, not compiled, not run on hardware
//
#include <hip/hip_runtime.h>

// ---------------- problem constants ----------------
#define BB 2
#define HH 16
#define LL 2048
#define DD 64
#define MM 256
#define BH (BB * HH)       // 32
#define CC 128             // chunk length
#define NCH (LL / CC)      // 16 chunks per (b,h)
#define ROWS (BH * LL)     // 65536 flattened rows

#define QSCALE 0.1767766952966369f   // 1024^-0.25
#define NCONST 0.0625f               // 256^-0.5
#define KEPS   1e-6f

typedef __bf16 bf16;
typedef __bf16 v16bf __attribute__((ext_vector_type(16)));
typedef __bf16 v8bf  __attribute__((ext_vector_type(8)));
typedef float  v8f   __attribute__((ext_vector_type(8)));

#define WMMA_BF16(a, b, c) \
  __builtin_amdgcn_wmma_f32_16x16x32_bf16(false, (a), false, (b), (short)0, (c), false, false)

__device__ inline v16bf cat8(v8bf lo, v8bf hi) {
  return __builtin_shufflevector(lo, hi, 0, 1, 2, 3, 4, 5, 6, 7, 8, 9, 10, 11, 12, 13, 14, 15);
}

// ------------- ordered-uint float max encoding -------------
__device__ inline unsigned ord_encode(float f) {
  unsigned b = __float_as_uint(f);
  return (b & 0x80000000u) ? ~b : (b | 0x80000000u);
}
__device__ inline float ord_decode(unsigned u) {
  unsigned b = (u & 0x80000000u) ? (u ^ 0x80000000u) : ~u;
  return __uint_as_float(b);
}

// ------------- fragment loaders (wave32, bf16 16x16x32) -------------
// A 16x32 (MxK), row-major source: lane's 16 values = two contiguous 8-elem runs
//   e<8  -> K = k0 + kb + e        (kb = (lane>>4)*8)
//   e>=8 -> K = k0 + kb + 16 + e-8
__device__ inline v16bf ldA_rm(const bf16* src, int ld, int row0, int k0, int lane) {
  const bf16* p = src + (size_t)(row0 + (lane & 15)) * ld + k0 + ((lane >> 4) << 3);
  v8bf lo = *(const v8bf*)(p);
  v8bf hi = *(const v8bf*)(p + 16);
  return cat8(lo, hi);
}

// Fragment-ready staged B: frag (kt,nt) occupies 512 elems; lane owns 16 contiguous
__device__ inline v16bf ldB_st(const bf16* base, int ntiles, int kt, int nt, int lane) {
  const bf16* p = base + ((size_t)(kt * ntiles + nt) << 9) + (lane << 4);
  v8bf lo = *(const v8bf*)(p);
  v8bf hi = *(const v8bf*)(p + 8);
  return cat8(lo, hi);
}
// staging index for logical B element (k, n); B is KxN, ntiles = N/16
__device__ inline int bstage_idx(int k, int n, int ntiles) {
  int kt = k >> 5, nt = n >> 4, kl = k & 31;
  int lane = (((kl >> 3) & 1) << 4) | (n & 15);
  return (((kt * ntiles + nt) << 9) + (lane << 4)) + (((kl >> 4) << 3) | (kl & 7));
}

// =============== K1: global max of h_of_k ===============
__global__ __launch_bounds__(256) void k_stab_max(const float* __restrict__ key,
                                                  unsigned* __restrict__ kstab_u) {
  __shared__ float smax[256];
  int row = blockIdx.x * 256 + threadIdx.x;
  const float* kr = key + (size_t)row * DD;
  float s = 0.f;
#pragma unroll
  for (int d = 0; d < DD; ++d) { float v = kr[d] * QSCALE; s += v * v; }
  smax[threadIdx.x] = -0.5f * s;
  __syncthreads();
  for (int off = 128; off > 0; off >>= 1) {
    if (threadIdx.x < off)
      smax[threadIdx.x] = fmaxf(smax[threadIdx.x], smax[threadIdx.x + off]);
    __syncthreads();
  }
  if (threadIdx.x == 0) atomicMax(kstab_u, ord_encode(smax[0]));
}

// =============== K2: featurize ===============
__global__ __launch_bounds__(256) void k_featurize(
    const float* __restrict__ qg, const float* __restrict__ kg,
    const float* __restrict__ rf, const unsigned* __restrict__ kstab_u,
    unsigned short* __restrict__ qp_o, unsigned short* __restrict__ kp_o) {
  __shared__ __align__(16) bf16 sRF[DD * MM];   // fragment-ready B (K=DD, N=MM) 32 KB
  __shared__ __align__(16) bf16 sQ[CC][DD];     // 16 KB row-major A
  __shared__ __align__(16) bf16 sK[CC][DD];     // 16 KB row-major A
  __shared__ float sHk[CC];
  int tid = threadIdx.x;
  size_t R0 = (size_t)blockIdx.x * CC;
  bf16* qp = (bf16*)qp_o;
  bf16* kp = (bf16*)kp_o;

  for (int i = tid; i < DD * MM; i += 256) {
    int d = i >> 8, m = i & 255;                 // rf row-major [DD][MM]
    sRF[bstage_idx(d, m, MM / 16)] = (bf16)rf[i];
  }
  for (int i = tid; i < CC * DD; i += 256) {
    int r = i >> 6, d = i & 63;
    (&sQ[0][0])[i] = (bf16)(qg[(R0 + r) * DD + d] * QSCALE);
    (&sK[0][0])[i] = (bf16)(kg[(R0 + r) * DD + d] * QSCALE);
  }
  if (tid < CC) {
    const float* kr = kg + (R0 + tid) * DD;
    float s = 0.f;
#pragma unroll
    for (int d = 0; d < DD; ++d) { float v = kr[d] * QSCALE; s += v * v; }
    sHk[tid] = -0.5f * s;
  }
  __syncthreads();

  float kstab = ord_decode(*kstab_u);
  int wave = tid >> 5, lane = tid & 31;
  int row0 = wave * 16;

  v16bf aQ0 = ldA_rm(&sQ[0][0], DD, row0, 0, lane);
  v16bf aQ1 = ldA_rm(&sQ[0][0], DD, row0, 32, lane);
  v16bf aK0 = ldA_rm(&sK[0][0], DD, row0, 0, lane);
  v16bf aK1 = ldA_rm(&sK[0][0], DD, row0, 32, lane);

  int col_l = lane & 15;
  int rb    = row0 + ((lane >> 4) << 3);
  for (int nt = 0; nt < MM / 16; ++nt) {
    v16bf b0 = ldB_st(sRF, MM / 16, 0, nt, lane);
    v16bf b1 = ldB_st(sRF, MM / 16, 1, nt, lane);
    v8f cq = {};
    cq = WMMA_BF16(aQ0, b0, cq);
    cq = WMMA_BF16(aQ1, b1, cq);
    v8f ck = {};
    ck = WMMA_BF16(aK0, b0, ck);
    ck = WMMA_BF16(aK1, b1, ck);
    int col = nt * 16 + col_l;
#pragma unroll
    for (int r = 0; r < 8; ++r) {
      int lrow = rb + r;
      qp[(R0 + lrow) * MM + col] = (bf16)(NCONST * (__expf(cq[r]) + KEPS));
      kp[(R0 + lrow) * MM + col] =
          (bf16)(NCONST * (__expf(sHk[lrow] + ck[r] - kstab) + KEPS));
    }
  }
}

// =============== K3: per-chunk Z_c = K'^T @ V  and per-chunk feature sums ===============
__global__ __launch_bounds__(256) void k_chunk_kv(
    const unsigned short* __restrict__ kp_i, const float* __restrict__ value,
    float* __restrict__ Z, float* __restrict__ KS) {
  __shared__ __align__(16) bf16 sKpT[MM][CC];   // 64 KB  K'^T row-major (A operand)
  __shared__ __align__(16) bf16 sVst[CC * DD];  // 16 KB  fragment-ready B (K=CC, N=DD)
  const bf16* kp = (const bf16*)kp_i;
  int tid = threadIdx.x;
  int bh = blockIdx.x / NCH, c = blockIdx.x % NCH;
  size_t rowbase = (size_t)bh * LL + (size_t)c * CC;

  for (int i = tid; i < CC * MM; i += 256) {
    int r = i >> 8, m = i & 255;
    sKpT[m][r] = kp[rowbase * MM + i];
  }
  for (int i = tid; i < CC * DD; i += 256) {
    int r = i >> 6, d = i & 63;
    sVst[bstage_idx(r, d, DD / 16)] = (bf16)value[rowbase * DD + i];
  }
  __syncthreads();

  { // per-chunk feature sums (contiguous now)
    float s = 0.f;
#pragma unroll 8
    for (int r = 0; r < CC; ++r) s += (float)sKpT[tid][r];
    KS[((size_t)bh * NCH + c) * MM + tid] = s;
  }

  int wave = tid >> 5, lane = tid & 31;
  float* Zc = Z + ((size_t)bh * NCH + c) * MM * DD;
#pragma unroll
  for (int mi = 0; mi < 2; ++mi) {
    int mt = wave * 2 + mi;
    v16bf a0 = ldA_rm(&sKpT[0][0], CC, mt * 16, 0,  lane);
    v16bf a1 = ldA_rm(&sKpT[0][0], CC, mt * 16, 32, lane);
    v16bf a2 = ldA_rm(&sKpT[0][0], CC, mt * 16, 64, lane);
    v16bf a3 = ldA_rm(&sKpT[0][0], CC, mt * 16, 96, lane);
#pragma unroll
    for (int nt = 0; nt < DD / 16; ++nt) {
      v8f acc = {};
      acc = WMMA_BF16(a0, ldB_st(sVst, DD / 16, 0, nt, lane), acc);
      acc = WMMA_BF16(a1, ldB_st(sVst, DD / 16, 1, nt, lane), acc);
      acc = WMMA_BF16(a2, ldB_st(sVst, DD / 16, 2, nt, lane), acc);
      acc = WMMA_BF16(a3, ldB_st(sVst, DD / 16, 3, nt, lane), acc);
      int n  = nt * 16 + (lane & 15);
      int rb = mt * 16 + ((lane >> 4) << 3);
#pragma unroll
      for (int r = 0; r < 8; ++r) Zc[(size_t)(rb + r) * DD + n] = acc[r];
    }
  }
}

// =============== K4: exclusive prefix over chunks (in-place) ===============
__global__ __launch_bounds__(256) void k_prefix(float* __restrict__ Z, float* __restrict__ KS) {
  int bh = blockIdx.x, tid = threadIdx.x;
  for (int e = tid; e < MM * DD; e += 256) {
    float run = 0.f;
    for (int c = 0; c < NCH; ++c) {
      size_t idx = ((size_t)bh * NCH + c) * (MM * DD) + e;
      float t = Z[idx]; Z[idx] = run; run += t;
    }
  }
  {
    float run = 0.f;
    for (int c = 0; c < NCH; ++c) {
      size_t idx = ((size_t)bh * NCH + c) * MM + tid;
      float t = KS[idx]; KS[idx] = run; run += t;
    }
  }
}

// =============== K5: O = tril(Q'K'^T)@V + Q'@S_prev, / denominator ===============
__global__ __launch_bounds__(256) void k_output(
    const unsigned short* __restrict__ qp_i, const unsigned short* __restrict__ kp_i,
    const float* __restrict__ value, const float* __restrict__ Z,
    const float* __restrict__ KS, float* __restrict__ out) {
  __shared__ __align__(16) bf16 sKpB[CC * MM];     // 64 KB frag-ready B (K=MM, N=CC)
  __shared__ __align__(16) bf16 sVst[CC * DD];     // 16 KB frag-ready B (K=CC, N=DD)
  __shared__ __align__(16) bf16 sSst[MM * DD];     // 32 KB frag-ready B (K=MM, N=DD)
  __shared__ float sSv[MM];                        // 1 KB
  __shared__ __align__(16) bf16 sP[8][16][CC];     // 32 KB masked scores (A, row-major)
  __shared__ float sDen[CC][2];
  const bf16* qp = (const bf16*)qp_i;
  const bf16* kp = (const bf16*)kp_i;
  int tid = threadIdx.x;
  int bh = blockIdx.x / NCH, c = blockIdx.x % NCH;
  size_t rowbase = (size_t)bh * LL + (size_t)c * CC;
  const float* Zc = Z + ((size_t)bh * NCH + c) * MM * DD;

  for (int i = tid; i < CC * MM; i += 256) {
    int row = i >> 8, m = i & 255;                 // B[k=m][n=row] = kp[row][m]
    sKpB[bstage_idx(m, row, CC / 16)] = kp[rowbase * MM + i];
  }
  for (int i = tid; i < CC * DD; i += 256) {
    int r = i >> 6, d = i & 63;
    sVst[bstage_idx(r, d, DD / 16)] = (bf16)value[rowbase * DD + i];
  }
  for (int i = tid; i < MM * DD; i += 256) {
    int k = i >> 6, d = i & 63;
    sSst[bstage_idx(k, d, DD / 16)] = (bf16)Zc[i];
  }
  if (tid < MM) sSv[tid] = KS[((size_t)bh * NCH + c) * MM + tid];
  __syncthreads();

  { // denominator inter part: Q'[row] . svec   (2 threads per row, vectorized)
    int row = tid >> 1, half = tid & 1;
    const bf16* qr = qp + (rowbase + row) * MM + half * (MM / 2);
    const float* sv = sSv + half * (MM / 2);
    float s = 0.f;
#pragma unroll
    for (int m0 = 0; m0 < MM / 2; m0 += 8) {
      v8bf x = *(const v8bf*)(qr + m0);
#pragma unroll
      for (int j = 0; j < 8; ++j) s += (float)x[j] * sv[m0 + j];
    }
    sDen[row][half] = s;
  }

  int wave = tid >> 5, lane = tid & 31;
  const bf16* qpG = qp + rowbase * MM;
  int row0 = wave * 16;

  // hoist Q' A-fragments (reused by scores and inter GEMM)
  v16bf qA[MM / 32];
#pragma unroll
  for (int ks = 0; ks < MM / 32; ++ks)
    qA[ks] = ldA_rm(qpG, MM, row0, ks * 32, lane);

  float rsum[8];
#pragma unroll
  for (int r = 0; r < 8; ++r) rsum[r] = 0.f;
  int col_l = lane & 15;
  int rb    = (lane >> 4) << 3;

  // scores P = tril(Q' K'^T); mask in C-frag; fold row-sums into denominator
  for (int nt = 0; nt < CC / 16; ++nt) {
    v8f p = {};
#pragma unroll
    for (int ks = 0; ks < MM / 32; ++ks)
      p = WMMA_BF16(qA[ks], ldB_st(sKpB, CC / 16, ks, nt, lane), p);
    int col = nt * 16 + col_l;
#pragma unroll
    for (int r = 0; r < 8; ++r) {
      int lrow = row0 + rb + r;
      float v  = (col <= lrow) ? p[r] : 0.f;
      rsum[r] += v;
      sP[wave][rb + r][col] = (bf16)v;
    }
  }
#pragma unroll
  for (int off = 1; off < 16; off <<= 1) {
#pragma unroll
    for (int r = 0; r < 8; ++r) rsum[r] += __shfl_xor(rsum[r], off, 32);
  }
  __syncthreads();

  // O = Q'@S_prev + P@V ; divide and store
  for (int nt = 0; nt < DD / 16; ++nt) {
    v8f o = {};
#pragma unroll
    for (int ks = 0; ks < MM / 32; ++ks)
      o = WMMA_BF16(qA[ks], ldB_st(sSst, DD / 16, ks, nt, lane), o);
#pragma unroll
    for (int ks = 0; ks < CC / 32; ++ks) {
      v16bf a = ldA_rm(&sP[wave][0][0], CC, 0, ks * 32, lane);
      o = WMMA_BF16(a, ldB_st(sVst, DD / 16, ks, nt, lane), o);
    }
    int n = nt * 16 + col_l;
#pragma unroll
    for (int r = 0; r < 8; ++r) {
      int lrow  = row0 + rb + r;
      float den = rsum[r] + sDen[lrow][0] + sDen[lrow][1];
      out[(rowbase + lrow) * DD + n] = o[r] / den;
    }
  }
}

// =============== host ===============
extern "C" void kernel_launch(void* const* d_in, const int* in_sizes, int n_in,
                              void* d_out, int out_size, void* d_ws, size_t ws_size,
                              hipStream_t stream) {
  const float* q  = (const float*)d_in[0];
  const float* k  = (const float*)d_in[1];
  const float* v  = (const float*)d_in[2];
  const float* rf = (const float*)d_in[3];
  float* out = (float*)d_out;

  char* ws = (char*)d_ws;
  const size_t QP_ELEMS = (size_t)ROWS * MM;                 // 16M bf16 elems
  unsigned*       kstab = (unsigned*)ws;
  unsigned short* qp    = (unsigned short*)(ws + 256);
  unsigned short* kp    = qp + QP_ELEMS;
  float*          Z     = (float*)(ws + 256 + 4 * QP_ELEMS); // after 2 bf16 buffers
  float*          KS    = Z + (size_t)BH * NCH * MM * DD;

  hipMemsetAsync(kstab, 0, 4, stream);  // ordered-uint 0 == minimal float

  k_stab_max <<<ROWS / 256, 256, 0, stream>>>(k, kstab);
  k_featurize<<<ROWS / CC,  256, 0, stream>>>(q, k, rf, kstab, qp, kp);
  k_chunk_kv <<<BH * NCH,   256, 0, stream>>>(kp, v, Z, KS);
  k_prefix   <<<BH,         256, 0, stream>>>(Z, KS);
  k_output   <<<BH * NCH,   256, 0, stream>>>(qp, kp, v, Z, KS, out);
}
